// MultiHeadAttention_4217657885032
// MI455X (gfx1250) — matrix operand order repaired and checked by reading, compiled, not run on hardware
//
#include <hip/hip_runtime.h>
#include <hip/hip_bf16.h>

typedef __attribute__((ext_vector_type(16))) _Float16 v16h;
typedef __attribute__((ext_vector_type(8)))  _Float16 v8h;
typedef __attribute__((ext_vector_type(4)))  _Float16 v4h;
typedef __attribute__((ext_vector_type(2)))  __fp16   v2fp;
typedef __attribute__((ext_vector_type(4)))  __fp16   v4fp;
typedef __attribute__((ext_vector_type(8)))  float    v8f;
typedef int v4i __attribute__((vector_size(16)));

constexpr int B_ = 4, S_ = 2048, D_ = 1024, H_ = 16, DH_ = 64;

// ---------------- async global->LDS (gfx1250) with safe fallback ----------------
#if defined(__has_builtin)
#  if __has_builtin(__builtin_amdgcn_global_load_async_to_lds_b128)
#    define HAS_ASYNC_LDS 1
#  endif
#endif
#ifndef HAS_ASYNC_LDS
#  define HAS_ASYNC_LDS 0
#endif

#if HAS_ASYNC_LDS
typedef __attribute__((address_space(1))) v4i gv4i;
typedef __attribute__((address_space(3))) v4i lv4i;
#endif

__device__ inline void copy16h_g2l(const _Float16* g, _Float16* l) {
#if HAS_ASYNC_LDS
  __builtin_amdgcn_global_load_async_to_lds_b128((gv4i*)g, (lv4i*)l, 0, 0);
  __builtin_amdgcn_global_load_async_to_lds_b128((gv4i*)(g + 8), (lv4i*)(l + 8), 0, 0);
#else
  *(v16h*)l = *(const v16h*)g;
#endif
}
__device__ inline void async_join() {
#if HAS_ASYNC_LDS
#  if __has_builtin(__builtin_amdgcn_s_wait_asynccnt)
  __builtin_amdgcn_s_wait_asynccnt(0);
#  else
  asm volatile("s_wait_asynccnt 0x0" ::: "memory");
#  endif
#endif
}

// pack 4 f32 -> 4 f16 (v_cvt_pk_rtz_f16_f32 x2), one 8-byte LDS store
__device__ inline void store_pk4(_Float16* dst, float4 v) {
  v2fp lo = __builtin_amdgcn_cvt_pkrtz(v.x, v.y);
  v2fp hi = __builtin_amdgcn_cvt_pkrtz(v.z, v.w);
  *(v4fp*)dst = __builtin_shufflevector(lo, hi, 0, 1, 2, 3);
}

// ---------------- GEMM config: C[M,N] = A[M,K] * W[N,K]^T + bias ----------------
constexpr int BM = 128, BN = 128, BK = 32;   // block tile, BK == WMMA K
constexpr int WMT = 4, WNT = 2;              // per-wave 64x32 tile = 4x2 subtiles
// 8 waves (256 threads) arranged 2 (M) x 4 (N)

__device__ inline v16h cat8(v8h lo, v8h hi) {
  return __builtin_shufflevector(lo, hi, 0,1,2,3,4,5,6,7,8,9,10,11,12,13,14,15);
}
__device__ inline v8f wmma_f16(v16h a, v16h b, v8f c) {
  return __builtin_amdgcn_wmma_f32_16x16x32_f16(false, a, false, b, (short)0, c,
                                                false, false);
}

// A-fragment (ISA 7.12.2, 16-bit A 16x32): lanes 0-15 row M=lane, K {0..7,16..23};
// lanes 16-31 row M=lane-16, K {8..15,24..31}.
__device__ inline v16h load_a_frag(const _Float16* base, int ld, int lane) {
  int row  = lane & 15;
  int koff = (lane >> 4) * 8;
  v8h lo = *(const v8h*)(base + row * ld + koff);
  v8h hi = *(const v8h*)(base + row * ld + koff + 16);
  return cat8(lo, hi);
}
// B-fragment from row-major [N][K] tile: lane picks row n, half-wave picks K group.
__device__ inline v16h load_b_frag(const _Float16* base, int ld, int lane) {
  int n  = lane & 15;
  int kb = (lane >> 4) * 8;
  v8h lo = *(const v8h*)(base + n * ld + kb);
  v8h hi = *(const v8h*)(base + n * ld + kb + 16);
  return cat8(lo, hi);
}

template <bool A_IS_F16, bool SPLIT_HEADS>
__global__ __launch_bounds__(256) void gemm_xwT_kernel(
    const void* __restrict__ Aptr,      // [M,K] f32 or f16
    const float* __restrict__ W,        // [N,K] f32
    const float* __restrict__ bias,     // [N]
    void* __restrict__ Out,             // f16 head-split or f32 row-major
    int M, int N, int K)
{
  __shared__ __align__(32) _Float16 sA[BM * BK];
  __shared__ __align__(32) _Float16 sB[BN * BK];

  const int tid  = threadIdx.x;
  const int wid  = tid >> 5;
  const int lane = tid & 31;
  const int wm   = wid >> 2;          // 0..1
  const int wn   = wid & 3;           // 0..3
  const int m0   = blockIdx.y * BM;
  const int n0   = blockIdx.x * BN;

  v8f acc[WMT][WNT] = {};

  for (int k0 = 0; k0 < K; k0 += BK) {
    // ---- stage A tile (128x32) into LDS as f16 ----
    if constexpr (A_IS_F16) {
      const _Float16* Af = (const _Float16*)Aptr;
      int r = tid >> 1, c = (tid & 1) * 16;
      copy16h_g2l(&Af[(size_t)(m0 + r) * K + k0 + c], &sA[r * BK + c]);
    } else {
      const float* Af = (const float*)Aptr;
      int rr = tid >> 3, cc = (tid & 7) * 4;
      #pragma unroll
      for (int i = 0; i < 4; ++i) {
        int r = rr + i * 32;
        float4 vv = *(const float4*)&Af[(size_t)(m0 + r) * K + k0 + cc];
        store_pk4(&sA[r * BK + cc], vv);
      }
      if (k0 + BK < K)
        __builtin_prefetch(&Af[(size_t)(m0 + rr) * K + k0 + BK + cc], 0, 0);
    }
    // ---- stage W tile (128x32) into LDS as f16 ----
    {
      int rr = tid >> 3, cc = (tid & 7) * 4;
      #pragma unroll
      for (int i = 0; i < 4; ++i) {
        int r = rr + i * 32;
        float4 vv = *(const float4*)&W[(size_t)(n0 + r) * K + k0 + cc];
        store_pk4(&sB[r * BK + cc], vv);
      }
      if (k0 + BK < K)
        __builtin_prefetch(&W[(size_t)(n0 + rr) * K + k0 + BK + cc], 0, 0);
    }
    if constexpr (A_IS_F16) async_join();
    __syncthreads();

    v16h afr[WMT];
    #pragma unroll
    for (int mt = 0; mt < WMT; ++mt)
      afr[mt] = load_a_frag(&sA[(wm * 64 + mt * 16) * BK], BK, lane);
    v16h bfr[WNT];
    #pragma unroll
    for (int nt = 0; nt < WNT; ++nt)
      bfr[nt] = load_b_frag(&sB[(wn * 32 + nt * 16) * BK], BK, lane);

    #pragma unroll
    for (int mt = 0; mt < WMT; ++mt)
      #pragma unroll
      for (int nt = 0; nt < WNT; ++nt)
        acc[mt][nt] = wmma_f16(afr[mt], bfr[nt], acc[mt][nt]);

    __syncthreads();
  }

  // ---- epilogue: C layout: VGPR r, lane l -> row r+(l/16)*8, col l&15 ----
  #pragma unroll
  for (int mt = 0; mt < WMT; ++mt) {
    #pragma unroll
    for (int nt = 0; nt < WNT; ++nt) {
      int col = n0 + wn * 32 + nt * 16 + (lane & 15);
      float bv = bias[col];
      #pragma unroll
      for (int r = 0; r < 8; ++r) {
        int row = m0 + wm * 64 + mt * 16 + r + (lane >> 4) * 8;
        float val = acc[mt][nt][r] + bv;
        if constexpr (SPLIT_HEADS) {
          int b = row / S_, s = row % S_;
          int h = col / DH_, d = col % DH_;
          ((_Float16*)Out)[(((size_t)b * H_ + h) * S_ + s) * DH_ + d] = (_Float16)val;
        } else {
          ((float*)Out)[(size_t)row * N + col] = val;
        }
      }
    }
  }
}

// ------- flash attention: one block (8 waves, 256 thr) per 128 q-rows per (b,h) ----
constexpr int QROWS = 128;  // q rows per block (16 per wave)

__global__ __launch_bounds__(256) void flash_attn_kernel(
    const _Float16* __restrict__ q,   // [B*H, S, 64]
    const _Float16* __restrict__ k,   // [B*H, S, 64]
    const _Float16* __restrict__ v,   // [B*H, S, 64]
    _Float16* __restrict__ ctx)       // [B, S, D] (heads re-interleaved)
{
  __shared__ __align__(32) _Float16 sK[64 * 64];      // [key][d]
  __shared__ __align__(32) _Float16 sVt[64 * 64];     // [d][key]  (transposed)
  __shared__ __align__(32) _Float16 sP[8 * 16 * 64];  // per-wave P tile [16][64]

  const int tid  = threadIdx.x;
  const int wid  = tid >> 5;
  const int lane = tid & 31;

  const int qblocks = S_ / QROWS;
  const int qb = blockIdx.x % qblocks;
  const int bh = blockIdx.x / qblocks;
  const int b  = bh / H_, h = bh % H_;

  const _Float16* qbase = q + ((size_t)bh * S_ + qb * QROWS) * DH_;
  const _Float16* kbase = k + (size_t)bh * S_ * DH_;
  const _Float16* vbase = v + (size_t)bh * S_ * DH_;

  // Q fragments for this wave's 16 rows; fold softmax scale (1/8, exact in f16)
  const int arow = wid * 16 + (lane & 15);
  const int koff = (lane >> 4) * 8;
  v16h qf[2];
  #pragma unroll
  for (int kk = 0; kk < 2; ++kk) {
    v8h lo = *(const v8h*)&qbase[(size_t)arow * DH_ + kk * 32 + koff];
    v8h hi = *(const v8h*)&qbase[(size_t)arow * DH_ + kk * 32 + koff + 16];
    v16h f = cat8(lo, hi);
    #pragma unroll
    for (int e = 0; e < 16; ++e) f[e] = f[e] * (_Float16)0.125f;
    qf[kk] = f;
  }

  float mrow[8], lrow[8];
  v8f o[4] = {};
  #pragma unroll
  for (int r = 0; r < 8; ++r) { mrow[r] = -3.0e38f; lrow[r] = 0.f; }

  const float log2e = 1.4426950408889634f;

  for (int j = 0; j < S_; j += 64) {
    __syncthreads();
    // ---- stage K [64][64]: 256 threads, 16 halves each (async -> LDS) ----
    {
      int r = tid >> 2, c = (tid & 3) * 16;
      copy16h_g2l(&kbase[(size_t)(j + r) * DH_ + c], &sK[r * 64 + c]);
    }
    // ---- stage V^T [64][64]: thread = 4 keys x 4 d's, packed b64 stores ----
    {
      int kg = (tid & 15) * 4;        // key group
      int dg = (tid >> 4) * 4;        // d group
      v4h vv[4];
      #pragma unroll
      for (int i = 0; i < 4; ++i)
        vv[i] = *(const v4h*)&vbase[(size_t)(j + kg + i) * DH_ + dg];
      #pragma unroll
      for (int d = 0; d < 4; ++d) {
        v4h col = { vv[0][d], vv[1][d], vv[2][d], vv[3][d] };
        *(v4h*)&sVt[(dg + d) * 64 + kg] = col;
      }
    }
    async_join();
    __syncthreads();

    // ---- scores S = (Q*sc) K^T : 4 tiles of 16 keys ----
    v8f s[4];
    #pragma unroll
    for (int nt = 0; nt < 4; ++nt) {
      v8f z = {};
      #pragma unroll
      for (int kk = 0; kk < 2; ++kk) {
        v16h bf = load_b_frag(&sK[(nt * 16) * 64 + kk * 32], 64, lane);
        z = wmma_f16(qf[kk], bf, z);
      }
      s[nt] = z;
    }

    // ---- online softmax (row in (VGPR r, half-wave); 16 cols across lanes) ----
    float fac[8], psum[8];
    #pragma unroll
    for (int r = 0; r < 8; ++r) {
      float mx = fmaxf(fmaxf(s[0][r], s[1][r]), fmaxf(s[2][r], s[3][r]));
      #pragma unroll
      for (int off = 1; off < 16; off <<= 1)
        mx = fmaxf(mx, __shfl_xor(mx, off, 16));
      float mnew = fmaxf(mrow[r], mx);
      fac[r] = exp2f((mrow[r] - mnew) * log2e);
      mrow[r] = mnew;
      psum[r] = 0.f;
    }
    #pragma unroll
    for (int nt = 0; nt < 4; ++nt) {
      #pragma unroll
      for (int r = 0; r < 8; ++r) {
        float p = exp2f((s[nt][r] - mrow[r]) * log2e);
        psum[r] += p;
        sP[wid * 1024 + (r + (lane >> 4) * 8) * 64 + nt * 16 + (lane & 15)] =
            (_Float16)p;
      }
    }
    #pragma unroll
    for (int r = 0; r < 8; ++r) {
      float ps = psum[r];
      #pragma unroll
      for (int off = 1; off < 16; off <<= 1) ps += __shfl_xor(ps, off, 16);
      lrow[r] = lrow[r] * fac[r] + ps;
    }
    #pragma unroll
    for (int nt = 0; nt < 4; ++nt)
      #pragma unroll
      for (int r = 0; r < 8; ++r) o[nt][r] *= fac[r];

    __syncthreads();

    // ---- O += P V : K-dim = 64 keys in 2 chunks ----
    #pragma unroll
    for (int kk = 0; kk < 2; ++kk) {
      v16h pf = load_a_frag(&sP[wid * 1024 + kk * 32], 64, lane);
      #pragma unroll
      for (int nt = 0; nt < 4; ++nt) {
        v16h bf = load_b_frag(&sVt[(nt * 16) * 64 + kk * 32], 64, lane);
        o[nt] = wmma_f16(pf, bf, o[nt]);
      }
    }
  }

  // ---- write context (f16), heads re-interleaved into [B,S,D] ----
  #pragma unroll
  for (int r = 0; r < 8; ++r) {
    int srow = qb * QROWS + wid * 16 + r + (lane >> 4) * 8;
    float rinv = __builtin_amdgcn_rcpf(lrow[r]);
    #pragma unroll
    for (int nt = 0; nt < 4; ++nt) {
      int d = h * DH_ + nt * 16 + (lane & 15);
      ctx[((size_t)b * S_ + srow) * D_ + d] = (_Float16)(o[nt][r] * rinv);
    }
  }
}

extern "C" void kernel_launch(void* const* d_in, const int* in_sizes, int n_in,
                              void* d_out, int out_size, void* d_ws, size_t ws_size,
                              hipStream_t stream) {
  const float* Q  = (const float*)d_in[0];
  const float* K  = (const float*)d_in[1];
  const float* V  = (const float*)d_in[2];
  const float* Wq = (const float*)d_in[3];
  const float* bq = (const float*)d_in[4];
  const float* Wk = (const float*)d_in[5];
  const float* bk = (const float*)d_in[6];
  const float* Wv = (const float*)d_in[7];
  const float* bv = (const float*)d_in[8];
  const float* Wo = (const float*)d_in[9];
  const float* bo = (const float*)d_in[10];

  const size_t elems = (size_t)B_ * S_ * D_;   // 8M halves per buffer
  _Float16* qbuf = (_Float16*)d_ws;
  _Float16* kbuf = qbuf + elems;
  _Float16* vbuf = kbuf + elems;
  _Float16* ctx  = vbuf + elems;

  const int M = B_ * S_;
  dim3 grid(D_ / BN, M / BM);   // (8, 64)

  gemm_xwT_kernel<false, true><<<grid, 256, 0, stream>>>(Q, Wq, bq, qbuf, M, D_, D_);
  gemm_xwT_kernel<false, true><<<grid, 256, 0, stream>>>(K, Wk, bk, kbuf, M, D_, D_);
  gemm_xwT_kernel<false, true><<<grid, 256, 0, stream>>>(V, Wv, bv, vbuf, M, D_, D_);

  flash_attn_kernel<<<B_ * H_ * (S_ / QROWS), 256, 0, stream>>>(qbuf, kbuf, vbuf, ctx);

  gemm_xwT_kernel<true, false><<<grid, 256, 0, stream>>>(ctx, Wo, bo, (float*)d_out,
                                                         M, D_, D_);
}
